// Model_48473000903511
// MI455X (gfx1250) — compile-verified, matrix operand order checked by reading, not run on hardware
//
#include <hip/hip_runtime.h>

// ---------------- problem constants (match reference) ----------------
#define N_NODES 50000
#define N_EDGES 500000
#define DIM 128
#define LAYERS 2
#define ND ((size_t)N_NODES * DIM)          // elements in one [N,D] tensor
#define DD (DIM * DIM)                      // one weight matrix

typedef __attribute__((ext_vector_type(16))) __bf16 v16bf;
typedef __attribute__((ext_vector_type(8)))  float  v8f;
typedef __attribute__((ext_vector_type(4)))  int    v4i;

#if defined(__has_builtin)
#if __has_builtin(__builtin_amdgcn_global_load_async_to_lds_b128)
#define HAVE_ASYNC_LDS 1
#endif
#endif

#if defined(HAVE_ASYNC_LDS)
typedef __attribute__((address_space(1))) v4i* gv4i_p;   // global int4*
typedef __attribute__((address_space(3))) v4i* lv4i_p;   // LDS int4*
#endif

__device__ __forceinline__ unsigned short f32_to_bf16(float f) {
  unsigned int u = __float_as_uint(f);
  u += 0x7FFFu + ((u >> 16) & 1u);          // round-to-nearest-even
  return (unsigned short)(u >> 16);
}

__device__ __forceinline__ unsigned int pack_bf16x2(float a, float b) {
  return (unsigned int)f32_to_bf16(a) | ((unsigned int)f32_to_bf16(b) << 16);
}

// ---------------- small utility kernels ----------------
__global__ void zero_f32_kernel(float* __restrict__ p, size_t n) {
  size_t i = (size_t)blockIdx.x * blockDim.x + threadIdx.x;
  if (i < n) p[i] = 0.0f;
}

__global__ void deg_kernel(const long long* __restrict__ dst,
                           float* __restrict__ deg, int nE) {
  int e = blockIdx.x * blockDim.x + threadIdx.x;
  if (e < nE) atomicAdd(&deg[(int)dst[e]], 1.0f);
}

__global__ void invdeg_kernel(float* __restrict__ deg, int n) {
  int i = blockIdx.x * blockDim.x + threadIdx.x;
  if (i < n) { float d = deg[i]; deg[i] = 1.0f / fmaxf(d, 1.0f); }
}

// Convert f32 weight matrices [*, 128, 128] to bf16 and TRANSPOSE each
// 128x128 matrix:  dst[m][n][k] = bf16(src[m][k][n]).
// The GEMM kernel then stages W^T into LDS as a straight contiguous copy.
__global__ void cvt_bf16_t_kernel(const float* __restrict__ src,
                                  unsigned short* __restrict__ dst, int n) {
  int i = blockIdx.x * blockDim.x + threadIdx.x;
  if (i < n) {
    int m = i / DD, rem = i - m * DD;
    int k = rem >> 7, nn = rem & 127;
    dst[(size_t)m * DD + nn * DIM + k] = f32_to_bf16(src[i]);
  }
}

// One float4 gather + 4 atomic f32 scatter-adds per (edge, 4-col slice).
__global__ void scatter_add_kernel(const long long* __restrict__ src,
                                   const long long* __restrict__ dst,
                                   const float* __restrict__ x,
                                   float* __restrict__ agg, int nE) {
  int tid = blockIdx.x * blockDim.x + threadIdx.x;
  int e = tid >> 5, g = tid & 31;
  if (e < nE) {
    int s = (int)src[e], d = (int)dst[e];
    float4 v = *reinterpret_cast<const float4*>(x + (size_t)s * DIM + g * 4);
    float* o = agg + (size_t)d * DIM + g * 4;
    atomicAdd(o + 0, v.x); atomicAdd(o + 1, v.y);
    atomicAdd(o + 2, v.z); atomicAdd(o + 3, v.w);
  }
}

// z = sigmoid(Pz) (in place); q = sigmoid(Pr) * h  (into Pr)
__global__ void zrq_kernel(float* __restrict__ Pz, float* __restrict__ Pr,
                           const float* __restrict__ h) {
  size_t i = (size_t)blockIdx.x * blockDim.x + threadIdx.x;
  if (i < ND) {
    float z = 1.0f / (1.0f + __expf(-Pz[i]));
    Pz[i] = z;
    float r = 1.0f / (1.0f + __expf(-Pr[i]));
    Pr[i] = r * h[i];
  }
}

// ---------------- fused multi-term WMMA GEMM ----------------
// out[M,128] = sum_t (rowscale_t ? diag(scale_t) : I) * A_t @ W_t  + bias0 + bias1
// mode 0: store pre-activation.
// mode 1: out = z*h + (1-z)*tanh(acc + ph1)   (GRU epilogue)
struct GemmTerm { const float* A; const float* scale; const unsigned short* Wt; };
struct GemmArgs {
  GemmTerm t[4];
  int nTerms;
  int mode;
  const float* bias0;
  const float* bias1;
  const float* ph1;   // mode 1
  const float* zbuf;  // mode 1
  const float* hbuf;  // mode 1
  float* out;
};

__launch_bounds__(128)
__global__ void sage_gemm_kernel(GemmArgs ga) {
  __shared__ unsigned short lds_a[64 * DIM];   // 16 KB: A strip, bf16
  __shared__ unsigned short lds_w[DIM * DIM];  // 32 KB: W^T [n][k], bf16
  const int tid    = threadIdx.x;
  const int lane   = tid & 31;
  const int wave   = tid >> 5;                 // 4 waves, 16 rows each
  const int laneHi = lane >> 4;
  const int lm     = lane & 15;
  const int rowBase = blockIdx.x * 64;

  v8f acc[8];
#pragma unroll
  for (int c = 0; c < 8; ++c)
#pragma unroll
    for (int j = 0; j < 8; ++j) acc[c][j] = 0.0f;

  for (int t = 0; t < ga.nTerms; ++t) {
    const float*          __restrict__ A  = ga.t[t].A;
    const float*          __restrict__ sc = ga.t[t].scale;
    const unsigned short* __restrict__ Wt = ga.t[t].Wt;

    // ---- stage W^T: straight 32 KB copy (pre-transposed in workspace) ----
#if defined(HAVE_ASYNC_LDS)
    {
      // Each lane DMAs 16 B per op; 128 threads x 16 ops cover 32 KB.
      const unsigned short* wsrc = Wt + tid * 128;
      unsigned short* wdst = lds_w + tid * 128;
#pragma unroll
      for (int ii = 0; ii < 16; ++ii) {
        __builtin_amdgcn_global_load_async_to_lds_b128(
            (gv4i_p)(wsrc + ii * 8), (lv4i_p)(wdst + ii * 8), 0, 0);
      }
    }
#else
    {
      const uint4* wsrc = reinterpret_cast<const uint4*>(Wt);
      uint4* wdst = reinterpret_cast<uint4*>(lds_w);
#pragma unroll
      for (int ii = 0; ii < 16; ++ii) wdst[tid * 16 + ii] = wsrc[tid * 16 + ii];
    }
#endif

    // ---- stage A: 64x128 f32 -> bf16 with optional row scale, b128 stores ----
    {
      const int r  = tid >> 1;
      const int cb = (tid & 1) * 64;
      const int gr = rowBase + r;
      const bool in = gr < N_NODES;
      float s = 1.0f;
      if (sc && in) s = sc[gr];
      const float4* __restrict__ ap =
          reinterpret_cast<const float4*>(A + (size_t)gr * DIM + cb);
      if (in) __builtin_prefetch(ap, 0, 0);
#pragma unroll
      for (int ii = 0; ii < 8; ++ii) {
        float4 v0 = make_float4(0.f, 0.f, 0.f, 0.f);
        float4 v1 = v0;
        if (in) { v0 = ap[ii * 2]; v1 = ap[ii * 2 + 1]; }
        uint4 p;
        p.x = pack_bf16x2(v0.x * s, v0.y * s);
        p.y = pack_bf16x2(v0.z * s, v0.w * s);
        p.z = pack_bf16x2(v1.x * s, v1.y * s);
        p.w = pack_bf16x2(v1.z * s, v1.w * s);
        *reinterpret_cast<uint4*>(lds_a + r * DIM + cb + ii * 8) = p;
      }
    }
#if defined(HAVE_ASYNC_LDS)
    asm volatile("s_wait_asynccnt 0x0" ::: "memory");
#endif
    __syncthreads();

    // ---- 4 K-steps x 8 col-tiles of v_wmma_f32_16x16x32_bf16 ----
    // 16-bit A/B fragment layout (ISA 7.12.2): lane m=lane&15; K-group
    // (k/8)&1 selects lane half; VGPR q holds K pair k = G*8 + (q&3)*2.
#pragma unroll
    for (int ks = 0; ks < 4; ++ks) {
      union { unsigned int u[8]; v16bf v; } af;
#pragma unroll
      for (int q = 0; q < 8; ++q) {
        int group = ((q >> 2) << 1) | laneHi;
        int k = ks * 32 + group * 8 + ((q & 3) << 1);
        af.u[q] = *reinterpret_cast<const unsigned int*>(
            lds_a + (wave * 16 + lm) * DIM + k);
      }
#pragma unroll
      for (int c = 0; c < 8; ++c) {
        union { unsigned int u[8]; v16bf v; } bfv;
#pragma unroll
        for (int q = 0; q < 8; ++q) {
          int group = ((q >> 2) << 1) | laneHi;
          int k = ks * 32 + group * 8 + ((q & 3) << 1);
          bfv.u[q] = *reinterpret_cast<const unsigned int*>(
              lds_w + (c * 16 + lm) * DIM + k);
        }
        acc[c] = __builtin_amdgcn_wmma_f32_16x16x32_bf16(
            false, af.v, false, bfv.v, (short)0, acc[c], false, false);
      }
    }
    __syncthreads();
  }

  // ---- epilogue: C/D layout — lanes 0-15: M=j, lanes 16-31: M=j+8; N=lane&15
  const int rw = rowBase + wave * 16 + 8 * laneHi;
#pragma unroll
  for (int c = 0; c < 8; ++c) {
    const int col = c * 16 + lm;
    float bv = 0.0f;
    if (ga.bias0) bv += ga.bias0[col];
    if (ga.bias1) bv += ga.bias1[col];
#pragma unroll
    for (int j = 0; j < 8; ++j) {
      const int row = rw + j;
      if (row < N_NODES) {
        const size_t idx = (size_t)row * DIM + col;
        float vv = acc[c][j] + bv;
        if (ga.mode == 0) {
          ga.out[idx] = vv;
        } else {
          float ht = tanhf(vv + ga.ph1[idx]);
          float z  = ga.zbuf[idx];
          ga.out[idx] = z * ga.hbuf[idx] + (1.0f - z) * ht;
        }
      }
    }
  }
}

// ---------------- host driver ----------------
static inline int cdiv(long long a, long long b) { return (int)((a + b - 1) / b); }

extern "C" void kernel_launch(void* const* d_in, const int* in_sizes, int n_in,
                              void* d_out, int out_size, void* d_ws, size_t ws_size,
                              hipStream_t stream) {
  (void)in_sizes; (void)n_in; (void)out_size; (void)ws_size;
  const float*      x   = (const float*)d_in[0];            // [N,D]
  const float*      h   = (const float*)d_in[1];            // [L,N,D]
  const long long*  ei  = (const long long*)d_in[2];        // [2,E] int64
  const float*      Wl  = (const float*)d_in[3];            // [L,6,D,D]
  const float*      Wr  = (const float*)d_in[4];            // [L,6,D,D]
  const float*      b   = (const float*)d_in[5];            // [L,6,D]
  float*            out = (float*)d_out;                    // [L,N,D]

  const long long* esrc = ei;
  const long long* edst = ei + N_EDGES;

  // workspace layout
  char* wsp = (char*)d_ws;
  auto alloc = [&](size_t bytes) {
    char* p = wsp; wsp += (bytes + 255) & ~(size_t)255; return p;
  };
  float* invdeg = (float*)alloc((size_t)N_NODES * 4);
  float* A1 = (float*)alloc(ND * 4);   // aggX, later aggQ
  float* A2 = (float*)alloc(ND * 4);   // aggH, later Ph1
  float* Pz = (float*)alloc(ND * 4);   // z pre-act -> z
  float* Pr = (float*)alloc(ND * 4);   // r pre-act -> q = r*h
  unsigned short* WlB = (unsigned short*)alloc((size_t)LAYERS * 6 * DD * 2);
  unsigned short* WrB = (unsigned short*)alloc((size_t)LAYERS * 6 * DD * 2);

  const int GEMM_BLOCKS = cdiv(N_NODES, 64);
  const int NW = LAYERS * 6 * DD;

  auto zero = [&](float* p, size_t n) {
    zero_f32_kernel<<<cdiv((long long)n, 256), 256, 0, stream>>>(p, n);
  };
  auto gemm = [&](int nT, GemmTerm t0, GemmTerm t1, GemmTerm t2, GemmTerm t3,
                  const float* b0, const float* b1, int mode,
                  const float* ph1, const float* zb, const float* hb, float* o) {
    GemmArgs ga{};
    ga.t[0] = t0; ga.t[1] = t1; ga.t[2] = t2; ga.t[3] = t3;
    ga.nTerms = nT; ga.mode = mode;
    ga.bias0 = b0; ga.bias1 = b1;
    ga.ph1 = ph1; ga.zbuf = zb; ga.hbuf = hb; ga.out = o;
    sage_gemm_kernel<<<GEMM_BLOCKS, 128, 0, stream>>>(ga);
  };

  // degrees -> inv_deg (reuse invdeg buffer for both)
  zero(invdeg, N_NODES);
  deg_kernel<<<cdiv(N_EDGES, 256), 256, 0, stream>>>(edst, invdeg, N_EDGES);
  invdeg_kernel<<<cdiv(N_NODES, 256), 256, 0, stream>>>(invdeg, N_NODES);

  // weights -> bf16, transposed per 128x128 matrix, once
  cvt_bf16_t_kernel<<<cdiv(NW, 256), 256, 0, stream>>>(Wl, WlB, NW);
  cvt_bf16_t_kernel<<<cdiv(NW, 256), 256, 0, stream>>>(Wr, WrB, NW);

  for (int i = 0; i < LAYERS; ++i) {
    const float* xin = (i == 0) ? x : out;       // layer 0 output feeds layer 1
    const float* hi  = h + (size_t)i * ND;
    const unsigned short* wl = WlB + (size_t)i * 6 * DD;
    const unsigned short* wr = WrB + (size_t)i * 6 * DD;
    const float* bi = b + (size_t)i * 6 * DIM;

    // aggregations of xin and h_i (shared across gates)
    zero(A1, ND); zero(A2, ND);
    scatter_add_kernel<<<cdiv((long long)N_EDGES * 32, 256), 256, 0, stream>>>(
        esrc, edst, xin, A1, N_EDGES);
    scatter_add_kernel<<<cdiv((long long)N_EDGES * 32, 256), 256, 0, stream>>>(
        esrc, edst, hi, A2, N_EDGES);

    // Pz = aggX@Wl0 + xin@Wr0 + aggH@Wl1 + h@Wr1 + b0 + b1
    gemm(4, {A1, invdeg, wl + 0 * DD}, {xin, nullptr, wr + 0 * DD},
            {A2, invdeg, wl + 1 * DD}, {hi, nullptr, wr + 1 * DD},
         bi + 0 * DIM, bi + 1 * DIM, 0, nullptr, nullptr, nullptr, Pz);
    // Pr = gates 2 (xin) + 3 (h)
    gemm(4, {A1, invdeg, wl + 2 * DD}, {xin, nullptr, wr + 2 * DD},
            {A2, invdeg, wl + 3 * DD}, {hi, nullptr, wr + 3 * DD},
         bi + 2 * DIM, bi + 3 * DIM, 0, nullptr, nullptr, nullptr, Pr);
    // Ph1 = gate 4 (xin part of h_tilde); aggH no longer needed -> reuse A2
    gemm(2, {A1, invdeg, wl + 4 * DD}, {xin, nullptr, wr + 4 * DD},
            {nullptr, nullptr, nullptr}, {nullptr, nullptr, nullptr},
         bi + 4 * DIM, nullptr, 0, nullptr, nullptr, nullptr, A2);

    // z = sigmoid(Pz); q = sigmoid(Pr) * h  (into Pr)
    zrq_kernel<<<cdiv((long long)ND, 256), 256, 0, stream>>>(Pz, Pr, hi);

    // aggregation of q (reuse A1)
    zero(A1, ND);
    scatter_add_kernel<<<cdiv((long long)N_EDGES * 32, 256), 256, 0, stream>>>(
        esrc, edst, Pr, A1, N_EDGES);

    // out_i = z*h + (1-z)*tanh(Ph1 + aggQ@Wl5 + q@Wr5 + b5)
    gemm(2, {A1, invdeg, wl + 5 * DD}, {Pr, nullptr, wr + 5 * DD},
            {nullptr, nullptr, nullptr}, {nullptr, nullptr, nullptr},
         bi + 5 * DIM, nullptr, 1, A2, Pz, hi, out + (size_t)i * ND);
  }
}
